// AttentionLayer_41377714929676
// MI455X (gfx1250) — matrix operand order repaired and checked by reading, not compiled, not run on hardware
//
#include <hip/hip_runtime.h>

// Problem constants (B=4, S=4096, D=128 from the reference).
#define BB   4
#define SS   4096
#define DD   128
#define NROW (BB * SS)   // 16384

typedef __attribute__((ext_vector_type(16))) _Float16 v16h;
typedef __attribute__((ext_vector_type(8)))  _Float16 v8h;
typedef __attribute__((ext_vector_type(8)))  float    v8f;

union V16H { v16h v; v8h h[2]; _Float16 e[16]; };

// ---------------------------------------------------------------------------
// Fragment loaders for V_WMMA_F32_16X16X32_F16 (ISA 7.12.2 layouts, wave32):
//  A (16x32 f16): lane L holds row m=L%16; halves 0..7  = K = (L/16)*8 + 0..7
//                                          halves 8..15 = K = (L/16)*8 + 16..23
//  B (32x16 f16): lane L holds col n=L%16; same K order as A
// Both reduce to contiguous 16B runs when the matrix is stored with the
// reduction dim contiguous — which is why V is stored transposed.
// ---------------------------------------------------------------------------
__device__ __forceinline__ v16h load_frag_b16(const _Float16* p) {
    V16H u;
    u.h[0] = *(const v8h*)(p);
    u.h[1] = *(const v8h*)(p + 16);
    return u.v;
}
__device__ __forceinline__ v16h load_frag_a16(const _Float16* p) {
    V16H u;
    u.h[0] = *(const v8h*)(p);        // K run 0..7   (relative)
    u.h[1] = *(const v8h*)(p + 16);   // K run 16..23 (relative)
    return u.v;
}

// ---------------------------------------------------------------------------
// Kernel 1: Q/K/V projections. Y = x @ W^T + b, output f16.
// One wave per 16x16 output tile; 4 WMMAs (k=32*4=128).
// mat 0 -> Qh[s][d], mat 1 -> Kh[s][d], mat 2 -> Vt[b][d][s] (transposed!)
// ---------------------------------------------------------------------------
__global__ __launch_bounds__(128) void proj_kernel(
    const float* __restrict__ x,
    const float* __restrict__ Wq, const float* __restrict__ bq,
    const float* __restrict__ Wk, const float* __restrict__ bk,
    const float* __restrict__ Wv, const float* __restrict__ bv,
    _Float16* __restrict__ Qh, _Float16* __restrict__ Kh,
    _Float16* __restrict__ Vt)
{
    const int wave = blockIdx.x * (blockDim.x >> 5) + (threadIdx.x >> 5);
    const int lane = threadIdx.x & 31;
    const int lo = lane & 15, hi = lane >> 4;

    const int stile = wave & 1023;         // 1024 row tiles of 16
    const int etile = (wave >> 10) & 7;    // 8 col tiles of 16
    const int mat   = wave >> 13;          // 0=Q, 1=K, 2=V

    const float* W;  const float* bias;
    if      (mat == 0) { W = Wq; bias = bq; }
    else if (mat == 1) { W = Wk; bias = bk; }
    else               { W = Wv; bias = bv; }

    const float* xr = x + (size_t)(stile * 16 + lo) * DD;   // A row (m = lo)
    const float* wr = W + (size_t)(etile * 16 + lo) * DD;   // B col (n = lo)

    v8f acc = {};
#pragma unroll
    for (int kc = 0; kc < 4; ++kc) {
        // A fragment from x (f32 -> f16), two 8-float runs
        V16H a;
        const float* pa = xr + kc * 32 + hi * 8;
#pragma unroll
        for (int i = 0; i < 8; ++i) a.e[i]     = (_Float16)pa[i];
#pragma unroll
        for (int i = 0; i < 8; ++i) a.e[i + 8] = (_Float16)pa[16 + i];
        // B fragment from W (f32 -> f16), two 8-float runs, same k order as A
        V16H b;
        const float* pb = wr + kc * 32 + hi * 8;
#pragma unroll
        for (int i = 0; i < 16; ++i) b.e[i] = (_Float16)pb[i < 8 ? i : i + 8];

        acc = __builtin_amdgcn_wmma_f32_16x16x32_f16(
                  false, a.v, false, b.v, (short)0, acc, false, false);
    }

    const float bn = bias[etile * 16 + lo];
#pragma unroll
    for (int r = 0; r < 8; ++r) {
        const int m  = r + hi * 8;                 // C row layout
        const int gr = stile * 16 + m;             // global row (b*S+s)
        const _Float16 val = (_Float16)(acc[r] + bn);
        if (mat == 2) {
            const int bb = gr >> 12, ss = gr & (SS - 1);
            Vt[(size_t)bb * DD * SS + (size_t)(etile * 16 + lo) * SS + ss] = val;
        } else {
            _Float16* O = (mat == 0) ? Qh : Kh;
            O[(size_t)gr * DD + etile * 16 + lo] = val;
        }
    }
}

// ---------------------------------------------------------------------------
// Kernel 2: fused flash attention with split-KV.
// One block per 16-row Q tile; ATTN_WAVES waves each handle a disjoint
// SS/ATTN_WAVES chunk of the KV sequence with an independent online softmax,
// then partials are merged through LDS:  O = sum_i exp(m_i - m*) O_i, etc.
// 1024 blocks * 8 waves = 8192 waves in flight.
// ---------------------------------------------------------------------------
#define ATTN_WAVES 8
#define KV_CHUNK   (SS / ATTN_WAVES)   // 512

__global__ __launch_bounds__(32 * ATTN_WAVES) void attn_kernel(
    const _Float16* __restrict__ Qh, const _Float16* __restrict__ Kh,
    const _Float16* __restrict__ Vt, float* __restrict__ out)
{
    __shared__ _Float16 lds_p[ATTN_WAVES][16 * 32];   // P staging (8 KB)
    __shared__ float    lds_o[ATTN_WAVES][16 * 128];  // partial O  (64 KB)
    __shared__ float    lds_m[ATTN_WAVES][16];
    __shared__ float    lds_l[ATTN_WAVES][16];
    __shared__ float    lds_f[ATTN_WAVES][16];        // merge factors

    const int wv   = threadIdx.x >> 5;
    const int lane = threadIdx.x & 31;
    const int lo = lane & 15, hi = lane >> 4;
    const int qt = blockIdx.x & 255;                  // q tile within batch
    const int b  = blockIdx.x >> 8;                   // batch

    const _Float16* Qb = Qh + (size_t)b * SS * DD;
    const _Float16* Kb = Kh + (size_t)b * SS * DD;
    const _Float16* Vb = Vt + (size_t)b * DD * SS;

    const float scale = 0.08838834764831845f;  // 1/sqrt(128), folded into Q

    // Q tile as 4 pre-scaled A fragments (held in registers whole kernel)
    v16h qf[4];
    {
        const _Float16* qr = Qb + (size_t)(qt * 16 + lo) * DD;
#pragma unroll
        for (int kc = 0; kc < 4; ++kc) {
            const _Float16* p = qr + kc * 32 + hi * 8;
            v8h h0 = *(const v8h*)(p);
            v8h h1 = *(const v8h*)(p + 16);
            V16H a;
#pragma unroll
            for (int i = 0; i < 8; ++i) a.e[i]     = (_Float16)((float)h0[i] * scale);
#pragma unroll
            for (int i = 0; i < 8; ++i) a.e[i + 8] = (_Float16)((float)h1[i] * scale);
            qf[kc] = a.v;
        }
    }

    v8f o[8] = {};                 // 16x128 f32 partial-output accumulator
    float mrow[8], lrow[8];        // per-row running max / sum (row = r + hi*8)
#pragma unroll
    for (int r = 0; r < 8; ++r) { mrow[r] = -1e30f; lrow[r] = 0.f; }

    _Float16* pl = lds_p[wv];
    const int kv0 = wv * KV_CHUNK;

    for (int kv = kv0; kv < kv0 + KV_CHUNK; kv += 32) {
        // ---- scores: two 16x16 tiles over 32 kv positions ----
        v8f s0 = {}, s1 = {};
#pragma unroll
        for (int kc = 0; kc < 4; ++kc) {
            v16h k0 = load_frag_b16(Kb + (size_t)(kv + lo) * DD + kc * 32 + hi * 8);
            s0 = __builtin_amdgcn_wmma_f32_16x16x32_f16(
                     false, qf[kc], false, k0, (short)0, s0, false, false);
            v16h k1 = load_frag_b16(Kb + (size_t)(kv + 16 + lo) * DD + kc * 32 + hi * 8);
            s1 = __builtin_amdgcn_wmma_f32_16x16x32_f16(
                     false, qf[kc], false, k1, (short)0, s1, false, false);
        }

        // ---- online softmax (row reductions within 16-lane halves) ----
#pragma unroll
        for (int r = 0; r < 8; ++r) {
            float mx = fmaxf(s0[r], s1[r]);
            mx = fmaxf(mx, __shfl_xor(mx, 1));
            mx = fmaxf(mx, __shfl_xor(mx, 2));
            mx = fmaxf(mx, __shfl_xor(mx, 4));
            mx = fmaxf(mx, __shfl_xor(mx, 8));
            const float mnew  = fmaxf(mrow[r], mx);
            const float alpha = __expf(mrow[r] - mnew);
            mrow[r] = mnew;
            const float p0 = __expf(s0[r] - mnew);
            const float p1 = __expf(s1[r] - mnew);
            float rs = p0 + p1;
            rs += __shfl_xor(rs, 1);
            rs += __shfl_xor(rs, 2);
            rs += __shfl_xor(rs, 4);
            rs += __shfl_xor(rs, 8);
            lrow[r] = lrow[r] * alpha + rs;
#pragma unroll
            for (int c = 0; c < 8; ++c) o[c][r] *= alpha;   // rescale O row
            const int m = r + hi * 8;                        // stash P tile
            pl[m * 32 + lo]      = (_Float16)p0;
            pl[m * 32 + lo + 16] = (_Float16)p1;
        }
        __asm__ volatile("s_wait_dscnt 0" ::: "memory");

        // ---- O += P(16x32) x V(32x128): P re-read in A layout from LDS ----
        v16h pf = load_frag_a16(pl + lo * 32 + hi * 8);
#pragma unroll
        for (int c = 0; c < 8; ++c) {
            v16h vf = load_frag_b16(Vb + (size_t)(c * 16 + lo) * SS + kv + hi * 8);
            o[c] = __builtin_amdgcn_wmma_f32_16x16x32_f16(
                       false, pf, false, vf, (short)0, o[c], false, false);
        }
    }

    // ---- publish partials to LDS ----
#pragma unroll
    for (int r = 0; r < 8; ++r) {
        const int m = r + hi * 8;
#pragma unroll
        for (int c = 0; c < 8; ++c)
            lds_o[wv][m * 128 + c * 16 + lo] = o[c][r];
        if (lo == 0) { lds_m[wv][m] = mrow[r]; lds_l[wv][m] = lrow[r]; }
    }
    __syncthreads();

    // ---- per-row merge factors: f_i = exp(m_i - m*) / sum_i exp(m_i-m*) l_i
    if (threadIdx.x < 16) {
        const int row = threadIdx.x;
        float mstar = -1e30f;
#pragma unroll
        for (int i = 0; i < ATTN_WAVES; ++i) mstar = fmaxf(mstar, lds_m[i][row]);
        float f[ATTN_WAVES];
        float lsum = 0.f;
#pragma unroll
        for (int i = 0; i < ATTN_WAVES; ++i) {
            f[i] = __expf(lds_m[i][row] - mstar);
            lsum += f[i] * lds_l[i][row];
        }
        const float inv = 1.0f / lsum;
#pragma unroll
        for (int i = 0; i < ATTN_WAVES; ++i) lds_f[i][row] = f[i] * inv;
    }
    __syncthreads();

    // ---- merge + write: thread t -> column d = t%128, rows (t/128)*8..+8 ----
    {
        const int d  = threadIdx.x & 127;
        const int r0 = (threadIdx.x >> 7) * 8;
        float* ob = out + ((size_t)b * SS + (size_t)qt * 16) * DD;
#pragma unroll
        for (int rr = 0; rr < 8; ++rr) {
            const int row = r0 + rr;
            float acc = 0.f;
#pragma unroll
            for (int i = 0; i < ATTN_WAVES; ++i)
                acc += lds_o[i][row * 128 + d] * lds_f[i][row];
            ob[(size_t)row * DD + d] = acc;
        }
    }
}

// ---------------------------------------------------------------------------
extern "C" void kernel_launch(void* const* d_in, const int* in_sizes, int n_in,
                              void* d_out, int out_size, void* d_ws, size_t ws_size,
                              hipStream_t stream) {
    const float* x  = (const float*)d_in[0];
    const float* Wq = (const float*)d_in[1];
    const float* bq = (const float*)d_in[2];
    const float* Wk = (const float*)d_in[3];
    const float* bk = (const float*)d_in[4];
    const float* Wv = (const float*)d_in[5];
    const float* bv = (const float*)d_in[6];
    float* out = (float*)d_out;

    // Workspace: Qh (4MB) + Kh (4MB) + Vt (4MB) in f16 = 12 MB.
    _Float16* Qh = (_Float16*)d_ws;
    _Float16* Kh = Qh + (size_t)NROW * DD;
    _Float16* Vt = Kh + (size_t)NROW * DD;

    // 3 mats * 8 etiles * 1024 stiles = 24576 waves; 4 waves/block.
    proj_kernel<<<24576 / 4, 128, 0, stream>>>(x, Wq, bq, Wk, bk, Wv, bv,
                                               Qh, Kh, Vt);
    // 1024 q-tiles (blocks); 8 split-KV waves per block.
    attn_kernel<<<BB * (SS / 16), 32 * ATTN_WAVES, 0, stream>>>(Qh, Kh, Vt, out);
}
